// FusedAttention_42348377539140
// MI455X (gfx1250) — matrix operand order repaired and checked by reading, not compiled, not run on hardware
//
#include <hip/hip_runtime.h>

typedef __bf16 bf16;
typedef __attribute__((ext_vector_type(16))) __bf16 v16bf;
typedef __attribute__((ext_vector_type(8)))  __bf16 v8bf;
typedef __attribute__((ext_vector_type(8)))  float  v8f;

// Problem constants (reference: B,S,D = 2,1024,2048 ; H,HKV,HD = 32,8,64)
constexpr int Bc   = 2;
constexpr int Sc   = 1024;
constexpr int Dc   = 2048;
constexpr int Hc   = 32;
constexpr int HKVc = 8;
constexpr int HDc  = 64;
constexpr int Mc   = Bc * Sc;      // 2048 rows
constexpr int NQ   = Hc * HDc;     // 2048
constexpr int NKV  = HKVc * HDc;   // 512

#define NEG_BIG (-1e30f)

static __device__ __forceinline__ v16bf cat8(v8bf lo, v8bf hi) {
  return __builtin_shufflevector(lo, hi, 0, 1, 2, 3, 4, 5, 6, 7,
                                         8, 9, 10, 11, 12, 13, 14, 15);
}

// ---------- fp32 -> bf16 convert ----------
__global__ __launch_bounds__(256) void k_cvt_bf16(const float* __restrict__ in,
                                                  bf16* __restrict__ out, int n) {
  int i = blockIdx.x * 256 + threadIdx.x;
  if (i < n) out[i] = (bf16)in[i];
}

// ---------- weight transpose+convert: in f32 [K][N] -> out bf16 [N][K] ----------
__global__ __launch_bounds__(256) void k_wtrans(const float* __restrict__ in,
                                                bf16* __restrict__ out, int K, int N) {
  int i = blockIdx.x * 256 + threadIdx.x;
  if (i >= K * N) return;
  int k = i % K;
  int n = i / K;
  out[i] = (bf16)in[(size_t)k * N + n];
}

// ---------- GEMM: C[M][N] f32 = A[M][K] bf16 (row major) x Bt[N][K] bf16 ----------
// 4 waves per block; each wave computes a 16(M) x 64(N) tile via wmma_f32_16x16x32_bf16.
__global__ __launch_bounds__(128) void k_gemm(const bf16* __restrict__ A,
                                              const bf16* __restrict__ Bt,
                                              float* __restrict__ C, int N, int K) {
  int lane = threadIdx.x & 31;
  int wave = threadIdx.x >> 5;
  int half = lane >> 4;
  int lm   = lane & 15;
  int m0 = blockIdx.y * 64 + wave * 16;
  int n0 = blockIdx.x * 64;
  int abase = half * 8;    // unused; A operand also uses kbase
  int kbase = half * 16;   // B operand per-lane K sub-offset
  const bf16* ar  = A  + (size_t)(m0 + lm) * K;
  const bf16* br0 = Bt + (size_t)(n0 + lm) * K + kbase;
  v8f acc0 = {}, acc1 = {}, acc2 = {}, acc3 = {};
  for (int k0 = 0; k0 < K; k0 += 32) {
    v16bf a  = *(const v16bf*)(ar + k0 + kbase);
    v16bf b0 = *(const v16bf*)(br0 + k0);
    v16bf b1 = *(const v16bf*)(br0 + (size_t)16 * K + k0);
    v16bf b2 = *(const v16bf*)(br0 + (size_t)32 * K + k0);
    v16bf b3 = *(const v16bf*)(br0 + (size_t)48 * K + k0);
    acc0 = __builtin_amdgcn_wmma_f32_16x16x32_bf16(false, a, false, b0, (short)0, acc0, false, false);
    acc1 = __builtin_amdgcn_wmma_f32_16x16x32_bf16(false, a, false, b1, (short)0, acc1, false, false);
    acc2 = __builtin_amdgcn_wmma_f32_16x16x32_bf16(false, a, false, b2, (short)0, acc2, false, false);
    acc3 = __builtin_amdgcn_wmma_f32_16x16x32_bf16(false, a, false, b3, (short)0, acc3, false, false);
  }
  float* cr = C + (size_t)(m0 + half * 8) * N + n0 + lm;
#pragma unroll
  for (int r = 0; r < 8; ++r) {
    cr[(size_t)r * N]      = acc0[r];
    cr[(size_t)r * N + 16] = acc1[r];
    cr[(size_t)r * N + 32] = acc2[r];
    cr[(size_t)r * N + 48] = acc3[r];
  }
}

// ---------- RoPE + convert + head-major transpose ----------
// in f32 [B][S][NH][HD]  ->  out bf16 [B][NH][S][HD]
__global__ __launch_bounds__(64) void k_rope(const float* __restrict__ in,
                                             const float* __restrict__ cosT,
                                             const float* __restrict__ sinT,
                                             bf16* __restrict__ out, int NH) {
  int idx = blockIdx.x;                 // b*S*NH + s*NH + h
  int h = idx % NH;
  int s = (idx / NH) % Sc;
  int b = idx / (NH * Sc);
  int i = threadIdx.x;                  // 0..63
  size_t base = (size_t)idx * HDc;
  float v  = in[base + i];
  float vr = (i < 32) ? -in[base + i + 32] : in[base + i - 32];
  float o  = v * cosT[s * HDc + i] + vr * sinT[s * HDc + i];
  out[(((size_t)b * NH + h) * Sc + s) * HDc + i] = (bf16)o;
}

// ---------- V transpose: f32 [B][S][HKV][HD] -> bf16 [B][HKV][HD][S] ----------
__global__ __launch_bounds__(256) void k_vtrans(const float* __restrict__ in,
                                                bf16* __restrict__ out) {
  int i = blockIdx.x * 256 + threadIdx.x;   // B*S*HKV*HD = 2^20
  int hd = i & 63;
  int g  = (i >> 6) & 7;
  int s  = (i >> 9) & 1023;
  int b  = i >> 19;
  out[(((size_t)b * HKVc + g) * HDc + hd) * Sc + s] = (bf16)in[i];
}

// ---------- Flash attention: one wave per (b, h, 16-row q tile) ----------
// Qh [B][H][S][HD] bf16, Kh [B][HKV][S][HD] bf16, Vt [B][HKV][HD][S] bf16
// Out [B][S][H*HD] bf16
__global__ __launch_bounds__(32) void k_attn(const bf16* __restrict__ Qh,
                                             const bf16* __restrict__ Kh,
                                             const bf16* __restrict__ Vt,
                                             bf16* __restrict__ Out) {
  __shared__ bf16 pl[16 * 32];            // P tile, C-layout -> A-layout staging
  int lane = threadIdx.x;
  int half = lane >> 4;
  int lm   = lane & 15;
  int qt = blockIdx.x & 63;               // S/16 = 64
  int h  = (blockIdx.x >> 6) & 31;
  int b  = blockIdx.x >> 11;
  int g  = h >> 2;                        // REP = 4
  int q0 = qt * 16;
  int abase = half * 8;
  int kbase = half * 16;

  // Q operand registers (16 rows x 64 K, two 16x32 steps)
  const bf16* qrow = Qh + (((size_t)b * Hc + h) * Sc + q0 + lm) * HDc;
  v16bf aq0 = *(const v16bf*)(qrow + kbase);
  v16bf aq1 = *(const v16bf*)(qrow + 32 + kbase);

  float m_i[8], l_i[8];
  v8f O[4] = {};
#pragma unroll
  for (int r = 0; r < 8; ++r) { m_i[r] = NEG_BIG; l_i[r] = 0.f; }

  const bf16* kp = Kh + ((size_t)b * HKVc + g) * Sc * HDc;
  const bf16* vp = Vt + ((size_t)b * HKVc + g) * HDc * Sc;

  for (int kt = 0; kt <= q0 + 15; kt += 32) {
    // scores: S[16 q][32 k] = Q(16x64) . K^T
    v8f s0 = {}, s1 = {};
    {
      const bf16* kr0 = kp + (size_t)(kt + lm) * HDc + kbase;
      const bf16* kr1 = kp + (size_t)(kt + 16 + lm) * HDc + kbase;
      v16bf b00 = *(const v16bf*)(kr0);
      v16bf b01 = *(const v16bf*)(kr0 + 32);
      v16bf b10 = *(const v16bf*)(kr1);
      v16bf b11 = *(const v16bf*)(kr1 + 32);
      s0 = __builtin_amdgcn_wmma_f32_16x16x32_bf16(false, aq0, false, b00, (short)0, s0, false, false);
      s0 = __builtin_amdgcn_wmma_f32_16x16x32_bf16(false, aq1, false, b01, (short)0, s0, false, false);
      s1 = __builtin_amdgcn_wmma_f32_16x16x32_bf16(false, aq0, false, b10, (short)0, s1, false, false);
      s1 = __builtin_amdgcn_wmma_f32_16x16x32_bf16(false, aq1, false, b11, (short)0, s1, false, false);
    }
    // scale + causal mask + online softmax (rows live in 16-lane halves)
    float p0[8], p1[8], mnew[8], rsum[8];
#pragma unroll
    for (int r = 0; r < 8; ++r) {
      int qp = q0 + half * 8 + r;
      float v0 = s0[r] * 0.125f;
      float v1 = s1[r] * 0.125f;
      if (kt + lm      > qp) v0 = NEG_BIG;
      if (kt + 16 + lm > qp) v1 = NEG_BIG;
      float rm = fmaxf(v0, v1);
      rm = fmaxf(rm, __shfl_xor(rm, 1));
      rm = fmaxf(rm, __shfl_xor(rm, 2));
      rm = fmaxf(rm, __shfl_xor(rm, 4));
      rm = fmaxf(rm, __shfl_xor(rm, 8));
      float mn = fmaxf(m_i[r], rm);
      mnew[r] = mn;
      p0[r] = __expf(v0 - mn);
      p1[r] = __expf(v1 - mn);
      float rs = p0[r] + p1[r];
      rs += __shfl_xor(rs, 1);
      rs += __shfl_xor(rs, 2);
      rs += __shfl_xor(rs, 4);
      rs += __shfl_xor(rs, 8);
      rsum[r] = rs;
    }
#pragma unroll
    for (int r = 0; r < 8; ++r) {
      float a = __expf(m_i[r] - mnew[r]);
      m_i[r] = mnew[r];
      l_i[r] = l_i[r] * a + rsum[r];
      O[0][r] *= a; O[1][r] *= a; O[2][r] *= a; O[3][r] *= a;
      pl[(half * 8 + r) * 32 + lm]      = (bf16)p0[r];
      pl[(half * 8 + r) * 32 + 16 + lm] = (bf16)p1[r];
    }
    __syncthreads();
    // P (16x32) as A operand, V tile (32 k x 64 hd) as B operand
    v16bf pa = *(const v16bf*)(&pl[lm * 32 + kbase]);
#pragma unroll
    for (int t = 0; t < 4; ++t) {
      const bf16* vr = vp + (size_t)(t * 16 + lm) * Sc + kt + kbase;
      v16bf vb = *(const v16bf*)vr;
      O[t] = __builtin_amdgcn_wmma_f32_16x16x32_bf16(false, pa, false, vb, (short)0, O[t], false, false);
    }
    __syncthreads();
  }

  // normalize + store bf16 to [B][S][H*HD]
  int row0 = q0 + half * 8;
#pragma unroll
  for (int r = 0; r < 8; ++r) {
    float inv = 1.0f / l_i[r];
    size_t base = ((size_t)b * Sc + row0 + r) * (size_t)NQ + (size_t)h * HDc + lm;
    Out[base]      = (bf16)(O[0][r] * inv);
    Out[base + 16] = (bf16)(O[1][r] * inv);
    Out[base + 32] = (bf16)(O[2][r] * inv);
    Out[base + 48] = (bf16)(O[3][r] * inv);
  }
}

extern "C" void kernel_launch(void* const* d_in, const int* in_sizes, int n_in,
                              void* d_out, int out_size, void* d_ws, size_t ws_size,
                              hipStream_t stream) {
  (void)in_sizes; (void)n_in; (void)out_size; (void)ws_size;
  const float* x    = (const float*)d_in[0];
  const float* wq   = (const float*)d_in[1];
  const float* wk   = (const float*)d_in[2];
  const float* wv   = (const float*)d_in[3];
  const float* wo   = (const float*)d_in[4];
  const float* cosT = (const float*)d_in[5];
  const float* sinT = (const float*)d_in[6];
  // d_in[7] = start_pos, fixed 0 in this problem (k_pos table only covers S keys)

  char* p = (char*)d_ws;
  size_t off = 0;
  auto take = [&](size_t bytes) -> char* {
    char* r = p + off;
    off = (off + bytes + 255) & ~(size_t)255;
    return r;
  };
  bf16*  x_bf = (bf16*)take((size_t)Mc * Dc * 2);
  bf16*  wq_t = (bf16*)take((size_t)NQ * Dc * 2);
  bf16*  wk_t = (bf16*)take((size_t)NKV * Dc * 2);
  bf16*  wv_t = (bf16*)take((size_t)NKV * Dc * 2);
  bf16*  wo_t = (bf16*)take((size_t)Dc * NQ * 2);
  float* q_f  = (float*)take((size_t)Mc * NQ * 4);
  float* k_f  = (float*)take((size_t)Mc * NKV * 4);
  float* v_f  = (float*)take((size_t)Mc * NKV * 4);
  bf16*  qh   = (bf16*)take((size_t)Bc * Hc * Sc * HDc * 2);
  bf16*  kh   = (bf16*)take((size_t)Bc * HKVc * Sc * HDc * 2);
  bf16*  vt   = (bf16*)take((size_t)Bc * HKVc * HDc * Sc * 2);
  bf16*  attn = (bf16*)q_f;   // q_f dead after RoPE; reuse for attention output

  // 1) convert activations / transpose+convert weights to bf16
  k_cvt_bf16<<<(Mc * Dc + 255) / 256, 256, 0, stream>>>(x, x_bf, Mc * Dc);
  k_wtrans<<<(Dc * NQ + 255) / 256, 256, 0, stream>>>(wq, wq_t, Dc, NQ);
  k_wtrans<<<(Dc * NKV + 255) / 256, 256, 0, stream>>>(wk, wk_t, Dc, NKV);
  k_wtrans<<<(Dc * NKV + 255) / 256, 256, 0, stream>>>(wv, wv_t, Dc, NKV);
  k_wtrans<<<(NQ * Dc + 255) / 256, 256, 0, stream>>>(wo, wo_t, NQ, Dc);

  // 2) QKV projections (WMMA bf16, f32 accumulate)
  k_gemm<<<dim3(NQ / 64, Mc / 64), 128, 0, stream>>>(x_bf, wq_t, q_f, NQ, Dc);
  k_gemm<<<dim3(NKV / 64, Mc / 64), 128, 0, stream>>>(x_bf, wk_t, k_f, NKV, Dc);
  k_gemm<<<dim3(NKV / 64, Mc / 64), 128, 0, stream>>>(x_bf, wv_t, v_f, NKV, Dc);

  // 3) RoPE (+ head-major layout) and V transpose
  k_rope<<<Bc * Sc * Hc, 64, 0, stream>>>(q_f, cosT, sinT, qh, Hc);
  k_rope<<<Bc * Sc * HKVc, 64, 0, stream>>>(k_f, cosT, sinT, kh, HKVc);
  k_vtrans<<<(Bc * Sc * HKVc * HDc) / 256, 256, 0, stream>>>(v_f, vt);

  // 4) causal GQA flash attention (WMMA for QK^T and PV)
  k_attn<<<Bc * Hc * (Sc / 16), 32, 0, stream>>>(qh, kh, vt, attn);

  // 5) output projection -> f32 result
  k_gemm<<<dim3(Dc / 64, Mc / 64), 128, 0, stream>>>(attn, wo_t, (float*)d_out, Dc, NQ);
}
